// MultiHeadAttention_4355096838417
// MI455X (gfx1250) — matrix operand order repaired and checked by reading, not compiled, not run on hardware
//
#include <hip/hip_runtime.h>

typedef __attribute__((ext_vector_type(16))) __bf16 v16bf;
typedef __attribute__((ext_vector_type(8)))  float  v8f;

#define D_MODEL 1024
#define DK      64
#define N_HEADS 16
#define SEQ     2048
#define BATCH   4

// ---- CDNA5 async global->LDS DMA (ASYNCcnt) with compile-safe fallback ----
#if defined(__AMDGCN__) && __has_builtin(__builtin_amdgcn_global_load_async_to_lds_b128) && \
    __has_builtin(__builtin_amdgcn_s_wait_asynccnt)
#define USE_ASYNC_LDS 1
#else
#define USE_ASYNC_LDS 0
#endif

#if USE_ASYNC_LDS
typedef int v4i_gcc __attribute__((vector_size(16)));
typedef __attribute__((address_space(1))) v4i_gcc* gptr_v4i;   // global (__device__) b128
typedef __attribute__((address_space(3))) v4i_gcc* lptr_v4i;   // LDS (__shared__) b128
#endif

__device__ __forceinline__ void copy_b128_to_lds(__bf16* lds_dst, const __bf16* g_src) {
#if USE_ASYNC_LDS
    __builtin_amdgcn_global_load_async_to_lds_b128(
        (gptr_v4i)g_src, (lptr_v4i)lds_dst, 0, 0);
#else
    *(uint4*)lds_dst = *(const uint4*)g_src;
#endif
}

__device__ __forceinline__ void wait_lds_copies() {
#if USE_ASYNC_LDS
    __builtin_amdgcn_s_wait_asynccnt(0);
#endif
}

// ---------------------------------------------------------------------------
// Kernel 1: QKV projection.  C(8192,1024) = X * W^T (W row-major (N,K)).
// Block = 256 threads (8 waves) -> 128x128 C tile; wave = 32x64 sub-tile.
// K-step 64 with register double-buffer on the fp32->bf16 staging path.
// Q,K stored (B,H,S,DK); V stored transposed (B,H,DK,S).
// ---------------------------------------------------------------------------
__global__ __launch_bounds__(256) void qkv_proj_kernel(
    const float* __restrict__ Xq, const float* __restrict__ Xk, const float* __restrict__ Xv,
    const float* __restrict__ Wq, const float* __restrict__ Wk, const float* __restrict__ Wv,
    __bf16* __restrict__ Qb, __bf16* __restrict__ Kb, __bf16* __restrict__ Vtb)
{
    __shared__ alignas(16) __bf16 Atile[128 * 64];   // [m][k]
    __shared__ alignas(16) __bf16 Btile[128 * 64];   // [n][k]

    const int which = blockIdx.z;
    const float* X = (which == 0) ? Xq : (which == 1) ? Xk : Xv;
    const float* W = (which == 0) ? Wq : (which == 1) ? Wk : Wv;

    const int m0   = blockIdx.x * 128;
    const int n0   = blockIdx.y * 128;
    const int tid  = threadIdx.x;
    const int lane = tid & 31;
    const int wave = tid >> 5;
    const int wm   = wave & 3;        // 4 M groups of 32 rows
    const int wn   = wave >> 2;       // 2 N groups of 64 cols
    const int hk   = (lane >> 4) * 8;
    const int kb2  = (lane >> 4) * 16;

    const int ldrow = tid >> 1;          // 0..127
    const int ldc0  = (tid & 1) * 32;    // 0 or 32

    const float* abase = X + (size_t)(m0 + ldrow) * D_MODEL + ldc0;
    const float* bbase = W + (size_t)(n0 + ldrow) * D_MODEL + ldc0;

    float areg[32], breg[32];
#pragma unroll
    for (int i = 0; i < 32; ++i) { areg[i] = abase[i]; breg[i] = bbase[i]; }

    v8f acc[2][4] = {};

    for (int k0 = 0; k0 < D_MODEL; k0 += 64) {
        __syncthreads();                 // previous compute done before overwrite
#pragma unroll
        for (int i = 0; i < 32; ++i) {
            Atile[ldrow * 64 + ldc0 + i] = (__bf16)areg[i];
            Btile[ldrow * 64 + ldc0 + i] = (__bf16)breg[i];
        }
        __syncthreads();
        if (k0 + 64 < D_MODEL) {         // pipeline: issue next tile's global loads
#pragma unroll
            for (int i = 0; i < 32; ++i) {
                areg[i] = abase[k0 + 64 + i];
                breg[i] = bbase[k0 + 64 + i];
            }
        }
#pragma unroll
        for (int ks = 0; ks < 2; ++ks) {
            v16bf am[2];
#pragma unroll
            for (int mi = 0; mi < 2; ++mi) {
                const int row = wm * 32 + mi * 16 + (lane & 15);
#pragma unroll
                for (int e = 0; e < 16; ++e)
                    am[mi][e] = Atile[row * 64 + ks * 32 + kb2 + e];
            }
#pragma unroll
            for (int nt = 0; nt < 4; ++nt) {
                v16bf b;
                const int col = wn * 64 + nt * 16 + (lane & 15);
#pragma unroll
                for (int e = 0; e < 16; ++e)
                    b[e] = Btile[col * 64 + ks * 32 + kb2 + e];
#pragma unroll
                for (int mi = 0; mi < 2; ++mi)
                    acc[mi][nt] = __builtin_amdgcn_wmma_f32_16x16x32_bf16(
                        false, am[mi], false, b, (short)0, acc[mi][nt], false, false);
            }
        }
    }

    const int rlo  = (lane >> 4) * 8;
    const int colb = lane & 15;
#pragma unroll
    for (int mi = 0; mi < 2; ++mi)
#pragma unroll
        for (int nt = 0; nt < 4; ++nt)
#pragma unroll
            for (int r = 0; r < 8; ++r) {
                const int m  = m0 + wm * 32 + mi * 16 + rlo + r;
                const int n  = n0 + wn * 64 + nt * 16 + colb;
                const int bb = m / SEQ, s = m % SEQ;
                const int hh = n / DK,  d = n % DK;
                const __bf16 val = (__bf16)acc[mi][nt][r];
                if (which == 0)
                    Qb[(((size_t)bb * N_HEADS + hh) * SEQ + s) * DK + d] = val;
                else if (which == 1)
                    Kb[(((size_t)bb * N_HEADS + hh) * SEQ + s) * DK + d] = val;
                else
                    Vtb[(((size_t)bb * N_HEADS + hh) * DK + d) * SEQ + s] = val;
            }
}

// ---------------------------------------------------------------------------
// Kernel 2: causal flash attention.  Block = 256 threads (8 waves) covering
// 128 query rows of one (b,h); wave owns 16 rows.  64-key KV tiles double-
// buffered in LDS via async global->LDS DMA; online softmax; QK^T and PV WMMA.
// ---------------------------------------------------------------------------
__global__ __launch_bounds__(256) void attn_kernel(
    const __bf16* __restrict__ Qb, const __bf16* __restrict__ Kb,
    const __bf16* __restrict__ Vtb, __bf16* __restrict__ Ob)
{
    __shared__ alignas(16) __bf16 Ktile[2][64 * 64];    // [kv][d]
    __shared__ alignas(16) __bf16 Vtile[2][64 * 64];    // [d][kv]
    __shared__ alignas(16) __bf16 Ptile[8][16 * 64];    // per-wave probs staging

    const int q0   = blockIdx.x * 128;
    const int h    = blockIdx.y;
    const int bb   = blockIdx.z;
    const int tid  = threadIdx.x;
    const int lane = tid & 31;
    const int wave = tid >> 5;

    const __bf16* Qh = Qb  + ((size_t)bb * N_HEADS + h) * (size_t)SEQ * DK;
    const __bf16* Kh = Kb  + ((size_t)bb * N_HEADS + h) * (size_t)SEQ * DK;
    const __bf16* Vh = Vtb + ((size_t)bb * N_HEADS + h) * (size_t)DK * SEQ;

    const int rlo  = (lane >> 4) * 8;
    const int coll = lane & 15;
    const int hk   = (lane >> 4) * 8;
    const int kb2  = (lane >> 4) * 16;

    // preload Q fragments (two K=32 slices over DK=64)
    v16bf qa[2];
    const int qrow = q0 + wave * 16 + coll;
#pragma unroll
    for (int ks = 0; ks < 2; ++ks)
#pragma unroll
        for (int e = 0; e < 16; ++e)
            qa[ks][e] = Qh[(size_t)qrow * DK + ks * 32 + kb2 + e];

    v8f oacc[4] = {};
    float mrow[8], lrow[8];
#pragma unroll
    for (int r = 0; r < 8; ++r) { mrow[r] = -1e30f; lrow[r] = 0.0f; }

    // stage one 64-key K tile (contiguous) + Vt tile (64 strided dk rows)
    auto stage = [&](int kv0, int buf) {
#pragma unroll
        for (int rep = 0; rep < 2; ++rep) {
            const int idx = tid + rep * 256;          // 512 x 16B chunks each
            copy_b128_to_lds(&Ktile[buf][idx * 8], Kh + (size_t)kv0 * DK + idx * 8);
            const int row = idx >> 3, part = idx & 7;
            copy_b128_to_lds(&Vtile[buf][row * 64 + part * 8],
                             Vh + (size_t)row * SEQ + kv0 + part * 8);
        }
    };

    const int nkv = q0 / 64 + 2;     // causal: kv0 <= q0 + 127
    stage(0, 0);
    for (int kt = 0; kt < nkv; ++kt) {
        const int buf = kt & 1;
        wait_lds_copies();           // our async chunks for tile kt done
        __syncthreads();             // all waves' chunks visible
        if (kt + 1 < nkv) stage((kt + 1) * 64, buf ^ 1);   // overlap with compute

        // ---- scores: four 16x16 tiles over the 64 keys
        float sc[4][8];
#pragma unroll
        for (int nt = 0; nt < 4; ++nt) {
            v8f c = {};
#pragma unroll
            for (int ks = 0; ks < 2; ++ks) {
                v16bf kbf;
#pragma unroll
                for (int e = 0; e < 16; ++e)
                    kbf[e] = Ktile[buf][(nt * 16 + coll) * 64 + ks * 32 + kb2 + e];
                c = __builtin_amdgcn_wmma_f32_16x16x32_bf16(
                    false, qa[ks], false, kbf, (short)0, c, false, false);
            }
#pragma unroll
            for (int r = 0; r < 8; ++r) {
                const int qr = q0 + wave * 16 + rlo + r;
                const int kc = kt * 64 + nt * 16 + coll;
                const float s = c[r] * 0.125f;            // 1/sqrt(64)
                sc[nt][r] = (kc <= qr) ? s : -1e30f;      // causal mask
            }
        }

        // ---- online softmax (row reductions inside each 16-lane half)
        float alpha[8];
#pragma unroll
        for (int r = 0; r < 8; ++r) {
            float mt = fmaxf(fmaxf(sc[0][r], sc[1][r]), fmaxf(sc[2][r], sc[3][r]));
#pragma unroll
            for (int off = 8; off >= 1; off >>= 1)
                mt = fmaxf(mt, __shfl_xor(mt, off, 32));
            const float mn = fmaxf(mrow[r], mt);
            alpha[r] = __expf(mrow[r] - mn);
            mrow[r]  = mn;
            float ps = 0.0f;
#pragma unroll
            for (int nt = 0; nt < 4; ++nt) {
                const float p = __expf(sc[nt][r] - mn);
                sc[nt][r] = p;
                ps += p;
            }
#pragma unroll
            for (int off = 8; off >= 1; off >>= 1)
                ps += __shfl_xor(ps, off, 32);
            lrow[r] = lrow[r] * alpha[r] + ps;
        }

        // ---- restage P (C layout -> A layout) through per-wave LDS
#pragma unroll
        for (int nt = 0; nt < 4; ++nt)
#pragma unroll
            for (int r = 0; r < 8; ++r)
                Ptile[wave][(rlo + r) * 64 + nt * 16 + coll] = (__bf16)sc[nt][r];
#pragma unroll
        for (int t = 0; t < 4; ++t)
#pragma unroll
            for (int r = 0; r < 8; ++r)
                oacc[t][r] *= alpha[r];
        __syncthreads();             // cross-lane LDS visibility

        v16bf pa[2];
#pragma unroll
        for (int ks = 0; ks < 2; ++ks)
#pragma unroll
            for (int e = 0; e < 16; ++e)
                pa[ks][e] = Ptile[wave][coll * 64 + ks * 32 + kb2 + e];

        // ---- PV: out(16 x DK) += P(16x64) * V(64xDK)
#pragma unroll
        for (int t = 0; t < 4; ++t)
#pragma unroll
            for (int ks = 0; ks < 2; ++ks) {
                v16bf vb;
#pragma unroll
                for (int e = 0; e < 16; ++e)
                    vb[e] = Vtile[buf][(t * 16 + coll) * 64 + ks * 32 + kb2 + e];
                oacc[t] = __builtin_amdgcn_wmma_f32_16x16x32_bf16(
                    false, pa[ks], false, vb, (short)0, oacc[t], false, false);
            }
    }

    // ---- normalize, store O as bf16 in (B, S, D_MODEL)
#pragma unroll
    for (int t = 0; t < 4; ++t)
#pragma unroll
        for (int r = 0; r < 8; ++r) {
            const float o  = oacc[t][r] / lrow[r];
            const int qr   = q0 + wave * 16 + rlo + r;
            const int dcol = t * 16 + coll;
            Ob[((size_t)bb * SEQ + qr) * D_MODEL + h * DK + dcol] = (__bf16)o;
        }
}

// ---------------------------------------------------------------------------
// Kernel 3: output projection  Y = O * Wp^T + bp (fp32 out).
// A tile (bf16) streamed by async DMA, double-buffered; B via reg pipeline.
// ---------------------------------------------------------------------------
__global__ __launch_bounds__(256) void out_proj_kernel(
    const __bf16* __restrict__ O, const float* __restrict__ Wp,
    const float* __restrict__ bp, float* __restrict__ Y)
{
    __shared__ alignas(16) __bf16 Atile[2][128 * 64];
    __shared__ alignas(16) __bf16 Btile[128 * 64];

    const int m0   = blockIdx.x * 128;
    const int n0   = blockIdx.y * 128;
    const int tid  = threadIdx.x;
    const int lane = tid & 31;
    const int wave = tid >> 5;
    const int wm   = wave & 3;
    const int wn   = wave >> 2;
    const int hk   = (lane >> 4) * 8;
    const int kb2  = (lane >> 4) * 16;

    const int ldrow = tid >> 1;
    const int ldc0  = (tid & 1) * 32;

    const float* bbase = Wp + (size_t)(n0 + ldrow) * D_MODEL + ldc0;
    float breg[32];
#pragma unroll
    for (int i = 0; i < 32; ++i) breg[i] = bbase[i];

    auto stageA = [&](int k0, int buf) {
        // 128 rows x 64 bf16 = 1024 x 16B chunks; 4 per thread
#pragma unroll
        for (int rep = 0; rep < 4; ++rep) {
            const int idx = tid + rep * 256;
            const int row = idx >> 3, part = idx & 7;
            copy_b128_to_lds(&Atile[buf][row * 64 + part * 8],
                             O + (size_t)(m0 + row) * D_MODEL + k0 + part * 8);
        }
    };
    stageA(0, 0);

    v8f acc[2][4] = {};

    for (int k0 = 0; k0 < D_MODEL; k0 += 64) {
        const int buf = (k0 >> 6) & 1;
        wait_lds_copies();
        __syncthreads();             // prev compute done; A(buf) visible
#pragma unroll
        for (int i = 0; i < 32; ++i)
            Btile[ldrow * 64 + ldc0 + i] = (__bf16)breg[i];
        if (k0 + 64 < D_MODEL) {
#pragma unroll
            for (int i = 0; i < 32; ++i) breg[i] = bbase[k0 + 64 + i];
            stageA(k0 + 64, buf ^ 1);
        }
        __syncthreads();             // B tile visible
#pragma unroll
        for (int ks = 0; ks < 2; ++ks) {
            v16bf am[2];
#pragma unroll
            for (int mi = 0; mi < 2; ++mi) {
                const int row = wm * 32 + mi * 16 + (lane & 15);
#pragma unroll
                for (int e = 0; e < 16; ++e)
                    am[mi][e] = Atile[buf][row * 64 + ks * 32 + kb2 + e];
            }
#pragma unroll
            for (int nt = 0; nt < 4; ++nt) {
                v16bf b;
                const int col = wn * 64 + nt * 16 + (lane & 15);
#pragma unroll
                for (int e = 0; e < 16; ++e)
                    b[e] = Btile[col * 64 + ks * 32 + kb2 + e];
#pragma unroll
                for (int mi = 0; mi < 2; ++mi)
                    acc[mi][nt] = __builtin_amdgcn_wmma_f32_16x16x32_bf16(
                        false, am[mi], false, b, (short)0, acc[mi][nt], false, false);
            }
        }
    }

    const int rlo  = (lane >> 4) * 8;
    const int colb = lane & 15;
#pragma unroll
    for (int mi = 0; mi < 2; ++mi)
#pragma unroll
        for (int nt = 0; nt < 4; ++nt)
#pragma unroll
            for (int r = 0; r < 8; ++r) {
                const int m = m0 + wm * 32 + mi * 16 + rlo + r;
                const int n = n0 + wn * 64 + nt * 16 + colb;
                Y[(size_t)m * D_MODEL + n] = acc[mi][nt][r] + bp[n];
            }
}

// ---------------------------------------------------------------------------
extern "C" void kernel_launch(void* const* d_in, const int* in_sizes, int n_in,
                              void* d_out, int out_size, void* d_ws, size_t ws_size,
                              hipStream_t stream)
{
    const float* Xq = (const float*)d_in[0];
    const float* Xk = (const float*)d_in[1];
    const float* Xv = (const float*)d_in[2];
    const float* Wq = (const float*)d_in[3];
    const float* Wk = (const float*)d_in[4];
    const float* Wv = (const float*)d_in[5];
    const float* Wp = (const float*)d_in[6];
    const float* bp = (const float*)d_in[7];
    float* Y = (float*)d_out;

    const size_t nElem = (size_t)BATCH * SEQ * D_MODEL;   // 8M elems
    __bf16* Qb  = (__bf16*)d_ws;
    __bf16* Kb  = Qb  + nElem;
    __bf16* Vtb = Kb  + nElem;
    __bf16* Ob  = Vtb + nElem;                            // total 64 MB bf16

    const dim3 blk(256);
    qkv_proj_kernel<<<dim3(BATCH * SEQ / 128, D_MODEL / 128, 3), blk, 0, stream>>>(
        Xq, Xk, Xv, Wq, Wk, Wv, Qb, Kb, Vtb);
    attn_kernel<<<dim3(SEQ / 128, N_HEADS, BATCH), blk, 0, stream>>>(Qb, Kb, Vtb, Ob);
    out_proj_kernel<<<dim3(BATCH * SEQ / 128, D_MODEL / 128), blk, 0, stream>>>(Ob, Wp, bp, Y);
}
